// NonOverlappingLocallyConnected1d_76081050681909
// MI455X (gfx1250) — matrix operand order repaired and checked by reading, not compiled, not run on hardware
//
#include <hip/hip_runtime.h>

// Problem constants (from reference)
#define B_DIM   64
#define IC      128
#define OC      128
#define OUT_DIM 1024
#define L       2048            // OUT_DIM * 2
#define KTOT    256             // IC * 2 (contraction length per output position)
#define XROW    264             // KTOT + 8 halfword pad (bank spread, keeps 16B align)
#define SCALE   0.0883883476483184f  // 1/sqrt(128)

typedef __attribute__((ext_vector_type(16))) __bf16        v16bf;
typedef __attribute__((ext_vector_type(8)))  float         v8f;
typedef __attribute__((ext_vector_type(4)))  unsigned int  v4u;

union Frag {
    v4u   q[2];   // two 16-byte chunks (ds_load_b128 each)
    v16bf v;      // WMMA operand view
};

__device__ __forceinline__ float bf16_trunc_f(float a) {
    unsigned int u = __builtin_bit_cast(unsigned int, a) & 0xFFFF0000u;
    return __builtin_bit_cast(float, u);
}
__device__ __forceinline__ unsigned int pack_bf16_pair(float a, float b) {
    unsigned int ua = __builtin_bit_cast(unsigned int, a);
    unsigned int ub = __builtin_bit_cast(unsigned int, b);
    return (ua >> 16) | (ub & 0xFFFF0000u);   // low half = first element
}

__global__ __launch_bounds__(256)
void nolc1d_wmma_bf16x3(const float* __restrict__ x,
                        const float* __restrict__ w,
                        float* __restrict__ out) {
    extern __shared__ __align__(16) unsigned short smem[];
    unsigned short* xh = smem;                         // [B_DIM][XROW] bf16 hi
    unsigned short* xl = xh + B_DIM * XROW;            // bf16 lo (residual)
    unsigned short* wh = xl + B_DIM * XROW;            // [OC][XROW]
    unsigned short* wl = wh + OC * XROW;

    const int tid = threadIdx.x;
    const int d   = blockIdx.x;

    // ---- Stage 1: gather strided float2 pairs for this d, split f32 -> bf16 hi/lo ----
    // x tile: B_DIM*IC = 8192 pairs -> 32 per thread
    for (int p = tid; p < B_DIM * IC; p += 256) {
        const int b = p >> 7;
        const int i = p & (IC - 1);
        const float2 v = *(const float2*)(x + (size_t)(b * IC + i) * L + 2 * d);
        const float h0 = bf16_trunc_f(v.x);
        const float h1 = bf16_trunc_f(v.y);
        *(unsigned int*)&xh[b * XROW + 2 * i] = pack_bf16_pair(v.x, v.y);
        *(unsigned int*)&xl[b * XROW + 2 * i] = pack_bf16_pair(v.x - h0, v.y - h1);
    }
    // w tile: OC*IC = 16384 pairs -> 64 per thread
    for (int p = tid; p < OC * IC; p += 256) {
        const int o = p >> 7;
        const int i = p & (IC - 1);
        const float2 v = *(const float2*)(w + (size_t)(o * IC + i) * L + 2 * d);
        const float h0 = bf16_trunc_f(v.x);
        const float h1 = bf16_trunc_f(v.y);
        *(unsigned int*)&wh[o * XROW + 2 * i] = pack_bf16_pair(v.x, v.y);
        *(unsigned int*)&wl[o * XROW + 2 * i] = pack_bf16_pair(v.x - h0, v.y - h1);
    }
    __syncthreads();

    // ---- Stage 2: per-wave WMMA. Wave wv owns N-tile wv (16 OC cols), 4 M-tiles. ----
    const int lane    = tid & 31;
    const int wv      = tid >> 5;
    const int lrow    = lane & 15;
    const int hi_half = lane >> 4;          // 0: lanes 0-15, 1: lanes 16-31
    const int n0      = wv * 16;

    v8f acc[4] = {};

    // Per ISA 7.12.2 (wave32):
    //  A 16x32 bf16: lane -> M=lrow; v0..3 hold K = kbase..kbase+7, v4..7 hold kbase+16..kbase+23,
    //                kbase = k0 + 8*hi_half.
//  B 32x16 bf16: lane -> N=lrow; v0..3 hold K = kbase..kbase+7, v4..7 hold kbase+16..kbase+23 (same as A).
    const unsigned short* wrow_h = wh + (n0 + lrow) * XROW;
    const unsigned short* wrow_l = wl + (n0 + lrow) * XROW;

    for (int k0 = 0; k0 < KTOT; k0 += 32) {
        Frag bhi, blo;
        const int kb = k0 + 8 * hi_half;
        bhi.q[0] = *(const v4u*)&wrow_h[kb];
        bhi.q[1] = *(const v4u*)&wrow_h[kb + 16];
        blo.q[0] = *(const v4u*)&wrow_l[kb];
        blo.q[1] = *(const v4u*)&wrow_l[kb + 16];

        const int ka = k0 + 8 * hi_half;
#pragma unroll
        for (int j = 0; j < 4; ++j) {
            const unsigned short* xrow_h = xh + (16 * j + lrow) * XROW;
            const unsigned short* xrow_l = xl + (16 * j + lrow) * XROW;
            Frag ahi, alo;
            ahi.q[0] = *(const v4u*)&xrow_h[ka];
            ahi.q[1] = *(const v4u*)&xrow_h[ka + 16];
            alo.q[0] = *(const v4u*)&xrow_l[ka];
            alo.q[1] = *(const v4u*)&xrow_l[ka + 16];

            // bf16x3 emulation of f32 product: hi*hi + hi*lo + lo*hi
            acc[j] = __builtin_amdgcn_wmma_f32_16x16x32_bf16(
                false, ahi.v, false, bhi.v, (short)0, acc[j], false, false);
            acc[j] = __builtin_amdgcn_wmma_f32_16x16x32_bf16(
                false, ahi.v, false, blo.v, (short)0, acc[j], false, false);
            acc[j] = __builtin_amdgcn_wmma_f32_16x16x32_bf16(
                false, alo.v, false, bhi.v, (short)0, acc[j], false, false);
        }
    }

    // ---- Epilogue: C/D layout (VGPR r: lanes 0-15 -> M=r, lanes 16-31 -> M=r+8) ----
    const int N = n0 + lrow;
#pragma unroll
    for (int j = 0; j < 4; ++j) {
        const int mbase = 16 * j + 8 * hi_half;
#pragma unroll
        for (int r = 0; r < 8; ++r) {
            const int M = mbase + r;
            out[((size_t)M * OC + N) * OUT_DIM + d] = acc[j][r] * SCALE;
        }
    }
}

extern "C" void kernel_launch(void* const* d_in, const int* in_sizes, int n_in,
                              void* d_out, int out_size, void* d_ws, size_t ws_size,
                              hipStream_t stream) {
    (void)in_sizes; (void)n_in; (void)out_size; (void)d_ws; (void)ws_size;
    const float* x = (const float*)d_in[0];
    const float* w = (const float*)d_in[1];
    float*     out = (float*)d_out;

    const size_t lds_bytes =
        (size_t)(2 * B_DIM * XROW + 2 * OC * XROW) * sizeof(unsigned short); // ~198 KB

    // Opt in to large dynamic LDS (320 KB per WGP on CDNA5); harmless if already allowed.
    (void)hipFuncSetAttribute((const void*)nolc1d_wmma_bf16x3,
                              hipFuncAttributeMaxDynamicSharedMemorySize,
                              (int)lds_bytes);

    nolc1d_wmma_bf16x3<<<dim3(OUT_DIM), dim3(256), lds_bytes, stream>>>(x, w, out);
}
